// ConvLayer_75239237091621
// MI455X (gfx1250) — matrix operand order repaired and checked by reading, not compiled, not run on hardware
//
#include <hip/hip_runtime.h>
#include <hip/hip_bf16.h>

// ---- problem constants -------------------------------------------------
#define MH   28
#define FHW  5
#define OHW  24
#define PP   576      // OH*OW
#define LL   25       // FH*FW
#define NN   64
#define MM   384
#define JIT  1e-6f
#define SU_STRIDE 408 // ushort stride for sU rows (816B: 16B aligned, conflict-friendly)

typedef __attribute__((ext_vector_type(16))) __bf16 bf16x16;
typedef __attribute__((ext_vector_type(8)))  float  f32x8;

struct Frag { union { bf16x16 v; uint4 q[2]; }; };

__device__ __forceinline__ unsigned short f2bf(float f) {
    unsigned int u = __float_as_uint(f);
    u += 0x7FFFu + ((u >> 16) & 1u);          // round-to-nearest-even
    return (unsigned short)(u >> 16);
}
__device__ __forceinline__ float bf2f(unsigned short h) {
    return __uint_as_float(((unsigned int)h) << 16);
}

// ---- setup kernels (one-time, O(M^3) total ~ 0.3 GFLOP) ----------------

// Xs = Z / ls, xnorm[m] = |Xs_m|^2
__global__ void k_prep_xs(const float* __restrict__ Z, const float* __restrict__ lsp,
                          float* __restrict__ Xs, float* __restrict__ xnorm) {
    int m = blockIdx.x * blockDim.x + threadIdx.x;
    if (m >= MM) return;
    float inv = 1.0f / *lsp;
    float s = 0.f;
    for (int l = 0; l < LL; ++l) {
        float v = Z[m * LL + l] * inv;
        Xs[m * LL + l] = v;
        s += v * v;
    }
    xnorm[m] = s;
}

// Kuu (+jitter) and augmented [Kuu | I] for Gauss-Jordan
__global__ void k_kuu_aug(const float* __restrict__ Xs, const float* __restrict__ xnorm,
                          const float* __restrict__ varp,
                          float* __restrict__ Kuu, float* __restrict__ aug) {
    int idx = blockIdx.x * 256 + threadIdx.x;
    if (idx >= MM * MM) return;
    int m = idx / MM, k = idx % MM;
    float dot = 0.f;
    for (int l = 0; l < LL; ++l) dot += Xs[m * LL + l] * Xs[k * LL + l];
    float d2 = fmaxf(xnorm[m] + xnorm[k] - 2.f * dot, 0.f);
    float v = (*varp) * expf(-0.5f * d2) + ((m == k) ? JIT : 0.f);
    Kuu[idx] = v;
    aug[m * 2 * MM + k] = v;
    aug[m * 2 * MM + MM + k] = (m == k) ? 1.f : 0.f;
}

// Gauss-Jordan inversion of SPD Kuu in aug[384][768]; one workgroup.
__global__ void k_invert(float* __restrict__ aug) {
    __shared__ float colk[MM];
    __shared__ float pivinv;
    const int t = threadIdx.x;
    for (int k = 0; k < MM; ++k) {
        for (int r = t; r < MM; r += 1024) colk[r] = aug[r * 2 * MM + k];
        if (t == 0) pivinv = 1.0f / aug[k * 2 * MM + k];
        __syncthreads();
        for (int c = t; c < 2 * MM; c += 1024) aug[k * 2 * MM + c] *= pivinv;
        __syncthreads();
        for (int i = t; i < MM * 2 * MM; i += 1024) {
            int r = i / (2 * MM), c = i % (2 * MM);
            if (r != k) aug[i] -= colk[r] * aug[k * 2 * MM + c];
        }
        __syncthreads();
    }
}

// B = tril(q_sqrt) tril(q_sqrt)^T - Kuu
__global__ void k_bmat(const float* __restrict__ qs, const float* __restrict__ Kuu,
                       float* __restrict__ B) {
    int idx = blockIdx.x * 256 + threadIdx.x;
    if (idx >= MM * MM) return;
    int m = idx / MM, k = idx % MM;
    int jm = (m < k) ? m : k;
    float s = 0.f;
    for (int j = 0; j <= jm; ++j) s += qs[m * MM + j] * qs[k * MM + j];
    B[idx] = s - Kuu[idx];
}

// T = Kinv @ B   (Kinv is right half of aug)
__global__ void k_tmat(const float* __restrict__ aug, const float* __restrict__ B,
                       float* __restrict__ T) {
    int idx = blockIdx.x * 256 + threadIdx.x;
    if (idx >= MM * MM) return;
    int m = idx / MM, k = idx % MM;
    float s = 0.f;
    for (int j = 0; j < MM; ++j) s += aug[m * 2 * MM + MM + j] * B[j * MM + k];
    T[idx] = s;
}

// C = T @ Kinv  -> bf16 ; cvec = Kinv @ q_mu
__global__ void k_cmat(const float* __restrict__ aug, const float* __restrict__ T,
                       const float* __restrict__ qmu,
                       unsigned short* __restrict__ Cbf, float* __restrict__ cvec) {
    int idx = blockIdx.x * 256 + threadIdx.x;
    if (idx >= MM * MM) return;
    int m = idx / MM, k = idx % MM;
    float s = 0.f;
    for (int j = 0; j < MM; ++j) s += T[m * MM + j] * aug[j * 2 * MM + MM + k];
    Cbf[idx] = f2bf(s);
    if (idx < MM) {
        float c = 0.f;
        for (int j = 0; j < MM; ++j) c += aug[idx * 2 * MM + MM + j] * qmu[j];
        cvec[idx] = c;
    }
}

// ---- main fused kernel: one workgroup per patch position p -------------
__global__ __launch_bounds__(256) void k_main(const float* __restrict__ X,
                                              const float* __restrict__ Xs,
                                              const float* __restrict__ xnorm,
                                              const float* __restrict__ varp,
                                              const float* __restrict__ lsp,
                                              const unsigned short* __restrict__ Cbf,
                                              const float* __restrict__ cvec,
                                              float* __restrict__ out) {
    __shared__ __align__(16) unsigned short sU[NN * SU_STRIDE]; // U^T in bf16: sU[n][m]
    __shared__ float sYs[NN * 26];
    __shared__ float sYn[NN];
    __shared__ float sVar[NN];

    const int tid = threadIdx.x;
    const int p = blockIdx.x;
    const int oh = p / OHW, ow = p % OHW;
    const float variance = *varp;
    const float inv_ls = 1.0f / *lsp;

    // stage scaled patches: sYs[n][l] = x[n, oh+fh, ow+fw] / ls
    for (int idx = tid; idx < NN * LL; idx += 256) {
        int n = idx / LL, l = idx % LL;
        int fh = l / FHW, fw = l % FHW;
        sYs[n * 26 + l] = X[n * (MH * MH) + (oh + fh) * MH + (ow + fw)] * inv_ls;
    }
    if (tid < NN) sVar[tid] = 0.0f;
    __syncthreads();
    if (tid < NN) {
        float s = 0.f;
        for (int l = 0; l < LL; ++l) { float v = sYs[tid * 26 + l]; s += v * v; }
        sYn[tid] = s;
    }
    __syncthreads();

    // U[m][n] = variance * exp(-0.5 * max(|Xs_m|^2 + |Ys_n|^2 - 2 Xs_m.Ys_n, 0))
    for (int idx = tid; idx < MM * NN; idx += 256) {
        int m = idx >> 6, n = idx & 63;
        const float* xr = Xs + m * LL;
        const float* yr = sYs + n * 26;
        float dot = 0.f;
#pragma unroll
        for (int l = 0; l < LL; ++l) dot += xr[l] * yr[l];
        float d2 = fmaxf(xnorm[m] + sYn[n] - 2.f * dot, 0.f);
        sU[n * SU_STRIDE + m] = f2bf(variance * expf(-0.5f * d2));
    }
    __syncthreads();

    // V = C @ U via WMMA bf16->f32; fused epilogue: sVar[n] += sum_m U[m][n]*V[m][n]
    const int lane = tid & 31;
    const int wave = tid >> 5;
    const int laneLo = (lane < 16);
    const int nsel = lane & 15;

    for (int mi = 0; mi < 3; ++mi) {
        const int m0 = (wave + mi * 8) * 16;
        // A fragment source: C rows m0..m0+15 (16-bit 16x32 A layout)
        const unsigned short* Arow = Cbf + (size_t)(m0 + nsel) * MM + (laneLo ? 0 : 8);
        for (int nt = 0; nt < 4; ++nt) {
            const int n0 = nt * 16;
            // B fragment source: U^T row = column n of U (32x16 B layout)
            const unsigned short* Brow = sU + (n0 + nsel) * SU_STRIDE + (laneLo ? 0 : 8);
            f32x8 acc = {};
#pragma unroll
            for (int kk = 0; kk < 12; ++kk) {
                const int k0 = kk * 32;
                Frag a, b;
                a.q[0] = *(const uint4*)(Arow + k0);
                a.q[1] = *(const uint4*)(Arow + k0 + 16);
                b.q[0] = *(const uint4*)(Brow + k0);
                b.q[1] = *(const uint4*)(Brow + k0 + 16);
                acc = __builtin_amdgcn_wmma_f32_16x16x32_bf16(
                    false, a.v, false, b.v, (short)0, acc, false, false);
            }
            // C/D layout: VGPR r, lane l -> M = m0 + r + (l<16?0:8), N = n0 + (l&15)
            const int n = n0 + nsel;
            const int mbase = m0 + (laneLo ? 0 : 8);
            uint4 uu = *(const uint4*)(sU + n * SU_STRIDE + mbase);
            const unsigned short* up = (const unsigned short*)&uu;
            float s = 0.f;
#pragma unroll
            for (int r = 0; r < 8; ++r) s += bf2f(up[r]) * acc[r];
            atomicAdd(&sVar[n], s);
        }
    }
    __syncthreads();

    // mean[n] = sum_k cvec[k] * U[k][n]; outputs NP-major
    if (tid < NN) {
        const int n = tid;
        const unsigned short* Un = sU + n * SU_STRIDE;
        float mean = 0.f;
        for (int k = 0; k < MM; ++k) mean += cvec[k] * bf2f(Un[k]);
        out[n * PP + p] = mean;
        out[NN * PP + n * PP + p] = variance + sVar[n];
    }
}

// ---- launcher ----------------------------------------------------------
extern "C" void kernel_launch(void* const* d_in, const int* in_sizes, int n_in,
                              void* d_out, int out_size, void* d_ws, size_t ws_size,
                              hipStream_t stream) {
    (void)in_sizes; (void)n_in; (void)out_size; (void)ws_size;
    const float* ND_X = (const float*)d_in[0];
    const float* Z    = (const float*)d_in[1];
    const float* qmu  = (const float*)d_in[2];
    const float* qs   = (const float*)d_in[3];
    const float* varp = (const float*)d_in[4];
    const float* lsp  = (const float*)d_in[5];
    float* out = (float*)d_out;

    float* w = (float*)d_ws;
    float* Kuu   = w;                         // 147456
    float* aug   = w + 147456;                // 294912  (384 x 768)
    float* Bm    = w + 147456 + 294912;       // 147456
    float* Tm    = Bm + 147456;               // 147456
    unsigned short* Cbf = (unsigned short*)(Tm + 147456); // 147456 ushorts (73728 f)
    float* cvec  = Tm + 147456 + 73728;       // 384
    float* Xs    = cvec + 384;                // 9600
    float* xnorm = Xs + 9600;                 // 384   (total ~3.2 MB)

    k_prep_xs<<<3, 128, 0, stream>>>(Z, lsp, Xs, xnorm);
    k_kuu_aug<<<(MM * MM + 255) / 256, 256, 0, stream>>>(Xs, xnorm, varp, Kuu, aug);
    k_invert<<<1, 1024, 0, stream>>>(aug);
    k_bmat<<<(MM * MM + 255) / 256, 256, 0, stream>>>(qs, Kuu, Bm);
    k_tmat<<<(MM * MM + 255) / 256, 256, 0, stream>>>(aug, Bm, Tm);
    k_cmat<<<(MM * MM + 255) / 256, 256, 0, stream>>>(aug, Tm, qmu, Cbf, cvec);
    k_main<<<PP, 256, 0, stream>>>(ND_X, Xs, xnorm, varp, lsp, Cbf, cvec, out);
}
